// LSTMRNN_55439437857105
// MI455X (gfx1250) — compile-verified, matrix operand order checked by reading, not run on hardware
//
#include <hip/hip_runtime.h>
#include <hip/hip_bf16.h>

#define H 1024
#define S_LEN 64
#define T_LEN 64
#define VOCAB 32000

typedef _Float16 v16h __attribute__((ext_vector_type(16)));
typedef _Float16 v8h  __attribute__((ext_vector_type(8)));
typedef float    v8f  __attribute__((ext_vector_type(8)));

__device__ __forceinline__ float sigm(float x) { return 1.0f / (1.0f + __expf(-x)); }

// ---------------------------------------------------------------------------
// K1: encoder LSTM, 1 block x 1024 threads, sequential over S steps.
// Writes enc_outs [S,H] and final h,c to workspace.
// ---------------------------------------------------------------------------
__global__ void k_encoder(const int* __restrict__ src_tokens,
                          const float* __restrict__ emb,    // [V,H]
                          const float* __restrict__ Wih,    // [4H,H]
                          const float* __restrict__ Whh,    // [4H,H]
                          const float* __restrict__ bih,    // [4H]
                          const float* __restrict__ bhh,    // [4H]
                          float* __restrict__ enc_outs,     // [S,H]
                          float* __restrict__ hc)           // [2H]
{
    __shared__ float xs[H];
    __shared__ float hs[H];
    __shared__ float cs[H];
    __shared__ float gates[4 * H];
    const int tid = threadIdx.x;

    hs[tid] = 0.0f;
    cs[tid] = 0.0f;
    __syncthreads();

    for (int s = 0; s < S_LEN; ++s) {
        const int tok = src_tokens[s];
        xs[tid] = emb[(size_t)tok * H + tid];
        __syncthreads();

        // 4 gate rows per thread
        #pragma unroll
        for (int r = 0; r < 4; ++r) {
            const int row = tid + r * H;
            float sum = bih[row] + bhh[row];
            const float4* wi = reinterpret_cast<const float4*>(Wih + (size_t)row * H);
            const float4* wh = reinterpret_cast<const float4*>(Whh + (size_t)row * H);
            for (int k4 = 0; k4 < H / 4; ++k4) {
                const float4 a = wi[k4];
                const float4 b = wh[k4];
                const int k = k4 * 4;
                sum += a.x * xs[k]     + a.y * xs[k + 1] + a.z * xs[k + 2] + a.w * xs[k + 3];
                sum += b.x * hs[k]     + b.y * hs[k + 1] + b.z * hs[k + 2] + b.w * hs[k + 3];
            }
            gates[row] = sum;
        }
        __syncthreads();

        // i, f, g, o  (torch gate order)
        const float gi = gates[tid];
        const float gf = gates[tid + H];
        const float gg = gates[tid + 2 * H];
        const float go = gates[tid + 3 * H];
        const float c2 = sigm(gf) * cs[tid] + sigm(gi) * tanhf(gg);
        const float h2 = sigm(go) * tanhf(c2);
        cs[tid] = c2;
        hs[tid] = h2;
        enc_outs[(size_t)s * H + tid] = h2;
        __syncthreads();
    }
    hc[tid]     = hs[tid];
    hc[tid + H] = cs[tid];
}

// ---------------------------------------------------------------------------
// K2: enc_proj[s, j] = att_bu[j] + sum_k enc_outs[s,k] * att_Wu[j,k]
// grid = S blocks, block = H threads
// ---------------------------------------------------------------------------
__global__ void k_encproj(const float* __restrict__ enc_outs, // [S,H]
                          const float* __restrict__ Wu,       // [H,H]
                          const float* __restrict__ bu,       // [H]
                          float* __restrict__ enc_proj)       // [S,H]
{
    const int s = blockIdx.x;
    const int j = threadIdx.x;
    const float4* wr = reinterpret_cast<const float4*>(Wu + (size_t)j * H);
    const float4* er = reinterpret_cast<const float4*>(enc_outs + (size_t)s * H);
    float sum = bu[j];
    for (int k4 = 0; k4 < H / 4; ++k4) {
        const float4 a = wr[k4];
        const float4 e = er[k4];
        sum += a.x * e.x + a.y * e.y + a.z * e.z + a.w * e.w;
    }
    enc_proj[(size_t)s * H + j] = sum;
}

// ---------------------------------------------------------------------------
// K3: decoder with additive attention, 1 block x 1024 threads, T steps.
// Produces pre[t,:] in f16 (A-matrix for the logits GEMM).
// ---------------------------------------------------------------------------
__global__ void k_decoder(const int* __restrict__ tgt_tokens,
                          const float* __restrict__ emb,      // dec_emb [V,H]
                          const float* __restrict__ We,       // [H,H]
                          const float* __restrict__ be,       // [H]
                          const float* __restrict__ av,       // att_v [H]
                          const float* __restrict__ w1W,      // [H,2H]
                          const float* __restrict__ w1b,      // [H]
                          const float* __restrict__ w2W,      // [H,2H]
                          const float* __restrict__ w2b,      // [H]
                          const float* __restrict__ Wih,      // [4H,H]
                          const float* __restrict__ Whh,      // [4H,H]
                          const float* __restrict__ bih,      // [4H]
                          const float* __restrict__ bhh,      // [4H]
                          const float* __restrict__ enc_outs, // [S,H]
                          const float* __restrict__ enc_proj, // [S,H]
                          const float* __restrict__ hc,       // [2H] initial
                          _Float16* __restrict__ preH)        // [T,H] f16 out
{
    __shared__ float hs[H];
    __shared__ float cs[H];
    __shared__ float wh[H];       // att_We@h + be, later reused as LSTM input
    __shared__ float red[H];      // energy partials
    __shared__ float tmp[2 * H];  // [context | x]
    __shared__ float gates[4 * H];
    __shared__ float attw[S_LEN];
    const int tid = threadIdx.x;

    hs[tid] = hc[tid];
    cs[tid] = hc[tid + H];
    __syncthreads();

    for (int t = 0; t < T_LEN; ++t) {
        const int tok = (t == 0) ? 0 : tgt_tokens[t - 1];  // BOS + teacher forcing

        // step 1: wh = We@h + be ; x into tmp[H..2H)
        {
            const float4* wr = reinterpret_cast<const float4*>(We + (size_t)tid * H);
            float sum = be[tid];
            for (int k4 = 0; k4 < H / 4; ++k4) {
                const float4 a = wr[k4];
                const int k = k4 * 4;
                sum += a.x * hs[k] + a.y * hs[k + 1] + a.z * hs[k + 2] + a.w * hs[k + 3];
            }
            wh[tid] = sum;
            tmp[H + tid] = emb[(size_t)tok * H + tid];
        }
        __syncthreads();

        // step 2: energy partials — 16 threads per s
        {
            const int s = tid >> 4;
            const int p = tid & 15;
            const float* ep = enc_proj + (size_t)s * H;
            float sum = 0.0f;
            const int k0 = p * 64;
            for (int k = k0; k < k0 + 64; ++k)
                sum += av[k] * tanhf(wh[k] + ep[k]);
            red[tid] = sum;
        }
        __syncthreads();

        // step 3: finish energies
        if (tid < S_LEN) {
            float e = 0.0f;
            #pragma unroll
            for (int p = 0; p < 16; ++p) e += red[tid * 16 + p];
            attw[tid] = e;
        }
        __syncthreads();

        // step 4: softmax (serial, trivial size)
        if (tid == 0) {
            float mx = -1e30f;
            for (int s = 0; s < S_LEN; ++s) mx = fmaxf(mx, attw[s]);
            float sum = 0.0f;
            for (int s = 0; s < S_LEN; ++s) { const float e = __expf(attw[s] - mx); attw[s] = e; sum += e; }
            const float inv = 1.0f / sum;
            for (int s = 0; s < S_LEN; ++s) attw[s] *= inv;
        }
        __syncthreads();

        // step 5: context into tmp[0..H)
        {
            float sum = 0.0f;
            for (int s = 0; s < S_LEN; ++s)
                sum += attw[s] * enc_outs[(size_t)s * H + tid];
            tmp[tid] = sum;
        }
        __syncthreads();

        // step 6: inp = w1W @ [context; x] + w1b  (reuse wh)
        {
            const float4* wr = reinterpret_cast<const float4*>(w1W + (size_t)tid * 2 * H);
            float sum = w1b[tid];
            for (int k4 = 0; k4 < (2 * H) / 4; ++k4) {
                const float4 a = wr[k4];
                const int k = k4 * 4;
                sum += a.x * tmp[k] + a.y * tmp[k + 1] + a.z * tmp[k + 2] + a.w * tmp[k + 3];
            }
            wh[tid] = sum;
        }
        __syncthreads();

        // step 7: LSTM gates
        #pragma unroll
        for (int r = 0; r < 4; ++r) {
            const int row = tid + r * H;
            float sum = bih[row] + bhh[row];
            const float4* wi = reinterpret_cast<const float4*>(Wih + (size_t)row * H);
            const float4* wv = reinterpret_cast<const float4*>(Whh + (size_t)row * H);
            for (int k4 = 0; k4 < H / 4; ++k4) {
                const float4 a = wi[k4];
                const float4 b = wv[k4];
                const int k = k4 * 4;
                sum += a.x * wh[k] + a.y * wh[k + 1] + a.z * wh[k + 2] + a.w * wh[k + 3];
                sum += b.x * hs[k] + b.y * hs[k + 1] + b.z * hs[k + 2] + b.w * hs[k + 3];
            }
            gates[row] = sum;
        }
        __syncthreads();

        // step 8: cell update
        {
            const float gi = gates[tid];
            const float gf = gates[tid + H];
            const float gg = gates[tid + 2 * H];
            const float go = gates[tid + 3 * H];
            const float c2 = sigm(gf) * cs[tid] + sigm(gi) * tanhf(gg);
            const float h2 = sigm(go) * tanhf(c2);
            cs[tid] = c2;
            hs[tid] = h2;
        }
        __syncthreads();

        // step 9: pre = w2W @ [h2; h2] + w2b   ->  f16 A-matrix row
        {
            const float4* wr = reinterpret_cast<const float4*>(w2W + (size_t)tid * 2 * H);
            float sum = w2b[tid];
            for (int k4 = 0; k4 < H / 4; ++k4) {
                const float4 a = wr[k4];
                const float4 b = wr[k4 + H / 4];
                const int k = k4 * 4;
                sum += (a.x + b.x) * hs[k]     + (a.y + b.y) * hs[k + 1]
                     + (a.z + b.z) * hs[k + 2] + (a.w + b.w) * hs[k + 3];
            }
            preH[(size_t)t * H + tid] = (_Float16)sum;
        }
        __syncthreads();
    }
}

// ---------------------------------------------------------------------------
// K4: logits[T,V] = Pre[T,H](f16) x out_W[V,H]^T (f16-converted) + out_b
// WMMA f32_16x16x32_f16. 8 waves/block; each wave owns TWO 16-column strips
// (reusing the four A fragments across both -> 8 WMMAs per K-chunk, halved
// A-matrix traffic). B (out_W) is streamed from HBM exactly once, coalesced.
// Fragment packing per CDNA5 ISA 7.12.2 (16-bit A/B 16x32, f32 C/D 16x16).
// ---------------------------------------------------------------------------
__global__ void k_logits(const _Float16* __restrict__ preA, // [64,1024]
                         const float* __restrict__ outW,    // [V,H]
                         const float* __restrict__ outb,    // [V]
                         float* __restrict__ out)           // [T,V]
{
    const int lane = threadIdx.x & 31;
    const int wave = threadIdx.x >> 5;
    const int lm   = lane & 15;   // row (A) / col (B) within tile
    const int hl   = lane >> 4;   // half selector
    const int n0   = blockIdx.x * 256 + wave * 32 + lm;   // first column strip
    const int n1   = n0 + 16;                             // second column strip

    v8f accA0 = {}, accA1 = {}, accA2 = {}, accA3 = {};   // tiles vs n0
    v8f accB0 = {}, accB1 = {}, accB2 = {}, accB3 = {};   // tiles vs n1
    const float* wrow0 = outW + (size_t)n0 * H;
    const float* wrow1 = outW + (size_t)n1 * H;

    for (int kc = 0; kc < H / 32; ++kc) {
        const int kb = kc * 32;

        // prefetch next K-chunk of both B rows (speculative, OOB-safe)
        __builtin_prefetch(wrow0 + kb + 32, 0, 0);
        __builtin_prefetch(wrow1 + kb + 32, 0, 0);

        // ---- B fragments: 16 f32 from out_W rows n0/n1, convert to f16 ----
        // element e<8  : K = kb + 8*hl + e
        // element e>=8 : K = kb + 16 + 8*hl + (e-8)
        v16h bf0, bf1;
        {
            const float4* p0 = reinterpret_cast<const float4*>(wrow0 + kb + 8 * hl);
            const float4* p1 = reinterpret_cast<const float4*>(wrow0 + kb + 16 + 8 * hl);
            const float4* q0 = reinterpret_cast<const float4*>(wrow1 + kb + 8 * hl);
            const float4* q1 = reinterpret_cast<const float4*>(wrow1 + kb + 16 + 8 * hl);
            const float4 c0 = p0[0], c1 = p0[1], c2 = p1[0], c3 = p1[1];
            const float4 d0 = q0[0], d1 = q0[1], d2 = q1[0], d3 = q1[1];
            bf0[0]  = (_Float16)c0.x; bf0[1]  = (_Float16)c0.y;
            bf0[2]  = (_Float16)c0.z; bf0[3]  = (_Float16)c0.w;
            bf0[4]  = (_Float16)c1.x; bf0[5]  = (_Float16)c1.y;
            bf0[6]  = (_Float16)c1.z; bf0[7]  = (_Float16)c1.w;
            bf0[8]  = (_Float16)c2.x; bf0[9]  = (_Float16)c2.y;
            bf0[10] = (_Float16)c2.z; bf0[11] = (_Float16)c2.w;
            bf0[12] = (_Float16)c3.x; bf0[13] = (_Float16)c3.y;
            bf0[14] = (_Float16)c3.z; bf0[15] = (_Float16)c3.w;
            bf1[0]  = (_Float16)d0.x; bf1[1]  = (_Float16)d0.y;
            bf1[2]  = (_Float16)d0.z; bf1[3]  = (_Float16)d0.w;
            bf1[4]  = (_Float16)d1.x; bf1[5]  = (_Float16)d1.y;
            bf1[6]  = (_Float16)d1.z; bf1[7]  = (_Float16)d1.w;
            bf1[8]  = (_Float16)d2.x; bf1[9]  = (_Float16)d2.y;
            bf1[10] = (_Float16)d2.z; bf1[11] = (_Float16)d2.w;
            bf1[12] = (_Float16)d3.x; bf1[13] = (_Float16)d3.y;
            bf1[14] = (_Float16)d3.z; bf1[15] = (_Float16)d3.w;
        }

        // ---- A fragments for the 4 M-tiles (f16, straight 16B loads) ----
        v16h af[4];
        #pragma unroll
        for (int mt = 0; mt < 4; ++mt) {
            const _Float16* arow = preA + (size_t)(mt * 16 + lm) * H + kb + 8 * hl;
            const v8h a0 = *reinterpret_cast<const v8h*>(arow);
            const v8h a1 = *reinterpret_cast<const v8h*>(arow + 16);
            #pragma unroll
            for (int i = 0; i < 8; ++i) { af[mt][i] = a0[i]; af[mt][8 + i] = a1[i]; }
        }

        accA0 = __builtin_amdgcn_wmma_f32_16x16x32_f16(false, af[0], false, bf0, (short)0, accA0, false, false);
        accA1 = __builtin_amdgcn_wmma_f32_16x16x32_f16(false, af[1], false, bf0, (short)0, accA1, false, false);
        accA2 = __builtin_amdgcn_wmma_f32_16x16x32_f16(false, af[2], false, bf0, (short)0, accA2, false, false);
        accA3 = __builtin_amdgcn_wmma_f32_16x16x32_f16(false, af[3], false, bf0, (short)0, accA3, false, false);
        accB0 = __builtin_amdgcn_wmma_f32_16x16x32_f16(false, af[0], false, bf1, (short)0, accB0, false, false);
        accB1 = __builtin_amdgcn_wmma_f32_16x16x32_f16(false, af[1], false, bf1, (short)0, accB1, false, false);
        accB2 = __builtin_amdgcn_wmma_f32_16x16x32_f16(false, af[2], false, bf1, (short)0, accB2, false, false);
        accB3 = __builtin_amdgcn_wmma_f32_16x16x32_f16(false, af[3], false, bf1, (short)0, accB3, false, false);
    }

    // ---- store: C/D layout — VGPR r holds M = r + 8*hl, N = lane%16 ----
    const float bias0 = outb[n0];
    const float bias1 = outb[n1];
    #pragma unroll
    for (int r = 0; r < 8; ++r) {
        const int m = r + 8 * hl;
        out[(size_t)(m)      * VOCAB + n0] = accA0[r] + bias0;
        out[(size_t)(m + 16) * VOCAB + n0] = accA1[r] + bias0;
        out[(size_t)(m + 32) * VOCAB + n0] = accA2[r] + bias0;
        out[(size_t)(m + 48) * VOCAB + n0] = accA3[r] + bias0;
        out[(size_t)(m)      * VOCAB + n1] = accB0[r] + bias1;
        out[(size_t)(m + 16) * VOCAB + n1] = accB1[r] + bias1;
        out[(size_t)(m + 32) * VOCAB + n1] = accB2[r] + bias1;
        out[(size_t)(m + 48) * VOCAB + n1] = accB3[r] + bias1;
    }
}

// ---------------------------------------------------------------------------
extern "C" void kernel_launch(void* const* d_in, const int* in_sizes, int n_in,
                              void* d_out, int out_size, void* d_ws, size_t ws_size,
                              hipStream_t stream) {
    const int*   src_tokens = (const int*)  d_in[0];
    const int*   tgt_tokens = (const int*)  d_in[1];
    // d_in[2] teacher_forcing_ratio == 1.0 (fixed teacher forcing)
    const float* enc_emb    = (const float*)d_in[3];
    const float* enc_Wih    = (const float*)d_in[4];
    const float* enc_Whh    = (const float*)d_in[5];
    const float* enc_bih    = (const float*)d_in[6];
    const float* enc_bhh    = (const float*)d_in[7];
    const float* dec_emb    = (const float*)d_in[8];
    const float* att_We     = (const float*)d_in[9];
    const float* att_be     = (const float*)d_in[10];
    const float* att_Wu     = (const float*)d_in[11];
    const float* att_bu     = (const float*)d_in[12];
    const float* att_v      = (const float*)d_in[13];
    const float* w1_W       = (const float*)d_in[14];
    const float* w1_b       = (const float*)d_in[15];
    const float* w2_W       = (const float*)d_in[16];
    const float* w2_b       = (const float*)d_in[17];
    const float* dec_Wih    = (const float*)d_in[18];
    const float* dec_Whh    = (const float*)d_in[19];
    const float* dec_bih    = (const float*)d_in[20];
    const float* dec_bhh    = (const float*)d_in[21];
    const float* out_W      = (const float*)d_in[22];
    const float* out_b      = (const float*)d_in[23];

    // workspace layout
    float* enc_outs = (float*)d_ws;                         // S*H
    float* enc_proj = enc_outs + S_LEN * H;                 // S*H
    float* hc       = enc_proj + S_LEN * H;                 // 2*H
    _Float16* preH  = (_Float16*)(hc + 2 * H);              // T*H f16

    k_encoder<<<1, H, 0, stream>>>(src_tokens, enc_emb, enc_Wih, enc_Whh,
                                   enc_bih, enc_bhh, enc_outs, hc);
    k_encproj<<<S_LEN, H, 0, stream>>>(enc_outs, att_Wu, att_bu, enc_proj);
    k_decoder<<<1, H, 0, stream>>>(tgt_tokens, dec_emb, att_We, att_be, att_v,
                                   w1_W, w1_b, w2_W, w2_b,
                                   dec_Wih, dec_Whh, dec_bih, dec_bhh,
                                   enc_outs, enc_proj, hc, preH);
    k_logits<<<VOCAB / 256, 256, 0, stream>>>(preH, out_W, out_b, (float*)d_out);
}
